// SelfAttention_28638841930062
// MI455X (gfx1250) — matrix operand order repaired and checked by reading, not compiled, not run on hardware
//
#include <hip/hip_runtime.h>
#include <hip/hip_bf16.h>

typedef __bf16 bhalf;
typedef __attribute__((ext_vector_type(16))) __bf16 v16bf;
typedef __attribute__((ext_vector_type(8)))  float  v8f;

#define N_FRAMES 4096
#define M_FEAT   1024
#define D_DIM    1024
#define APER     256
#define QSCALE   0.06f

#define TILE  64
#define KSTEP 32
#define LDSP  40   // 32 + 8 pad elements; 80B row stride keeps 16B alignment

__device__ __forceinline__ int imin(int a, int b) { return a < b ? a : b; }
__device__ __forceinline__ int imax(int a, int b) { return a > b ? a : b; }

// CDNA5 async DMA: global (per-lane 64-bit addr) -> LDS (per-lane 32-bit offset),
// 16 bytes per lane, tracked by ASYNCcnt (no VGPR round trip, no LOADcnt stall).
__device__ __forceinline__ void async_copy_b128(void* lds, const void* gptr) {
  unsigned loff = (unsigned)(unsigned long long)lds;       // low 32b of generic = LDS offset
  unsigned long long ga = (unsigned long long)gptr;
  asm volatile("global_load_async_to_lds_b128 %0, %1, off"
               :: "v"(loff), "v"(ga) : "memory");
}

// ---------------------------------------------------------------------------
// f32 -> bf16 conversion (row-major copy, used for x)
// ---------------------------------------------------------------------------
__global__ void cvt_f32_bf16_kernel(const float* __restrict__ in,
                                    bhalf* __restrict__ out, int n) {
  int i = blockIdx.x * blockDim.x + threadIdx.x;
  if (i < n) out[i] = (bhalf)in[i];
}

// ---------------------------------------------------------------------------
// f32 (K x N) -> bf16 transposed (N x K), LDS-tiled 32x32
// ---------------------------------------------------------------------------
__global__ __launch_bounds__(256)
void cvt_transpose_bf16_kernel(const float* __restrict__ in,
                               bhalf* __restrict__ out, int K, int N) {
  __shared__ bhalf t[32][33];
  const int bk = blockIdx.y * 32, bn = blockIdx.x * 32;
  const int tx = threadIdx.x & 31, ty = threadIdx.x >> 5;   // 32 x 8
#pragma unroll
  for (int i = 0; i < 32; i += 8)
    t[ty + i][tx] = (bhalf)in[(size_t)(bk + ty + i) * N + bn + tx];
  __syncthreads();
#pragma unroll
  for (int i = 0; i < 32; i += 8)
    out[(size_t)(bn + ty + i) * K + bk + tx] = t[tx][ty + i];
}

// ---------------------------------------------------------------------------
// Unified NT GEMM:  C = scale * A(MxK) @ Bt(NxK)^T  [+ bias]
//   - Cf  (f32), Cb (bf16), CbT (bf16, transposed with leading dim ldT): any subset
//   - band_mask > 0: logits mode - skip tiles wholly outside |i-j| < band_mask,
//     write -inf on masked elements
//   - band_k > 0: restrict K-loop to the band around row_base (attT @ V)
// 128 threads = 4 waves (2x2); each wave: 2x2 v_wmma_f32_16x16x32_bf16 accums.
// Double-buffered LDS tiles filled by global_load_async_to_lds_b128.
// ---------------------------------------------------------------------------
__global__ __launch_bounds__(128)
void gemm_nt_bf16_async_kernel(const bhalf* __restrict__ A,
                               const bhalf* __restrict__ Bt,
                               const float* __restrict__ bias,
                               float* __restrict__ Cf, bhalf* __restrict__ Cb,
                               bhalf* __restrict__ CbT, long ldT,
                               int Ndim, int Kdim, float scale,
                               int band_mask, int band_k) {
  const int row_base = blockIdx.y * TILE;
  const int col_base = blockIdx.x * TILE;
  if (band_mask > 0) {
    if (col_base > row_base + TILE - 1 + band_mask - 1 ||
        row_base > col_base + TILE - 1 + band_mask - 1)
      return;  // tile entirely masked; softmax writes the zeros
  }

  __shared__ bhalf As[2][TILE][LDSP];   // [buf][row][k]
  __shared__ bhalf Bs[2][TILE][LDSP];   // [buf][col][k]

  const int tid  = threadIdx.x;
  const int wave = tid >> 5;
  const int lane = tid & 31;
  const int wm = wave >> 1, wn = wave & 1;

  int k_lo = 0, k_hi = Kdim;
  if (band_k > 0) {
    k_lo = imax(row_base - (band_k - 1), 0) & ~(KSTEP - 1);
    k_hi = imin((row_base + TILE - 1 + band_k + KSTEP - 1) & ~(KSTEP - 1), Kdim);
  }
  const int nk = (k_hi - k_lo) / KSTEP;

  // per-thread chunk: rows r0 and r0+32, 8 bf16 (16B) starting at column c0
  const int r0 = tid >> 2;
  const int c0 = (tid & 3) << 3;

  auto issue = [&](int buf, int k0) {
    async_copy_b128(&As[buf][r0][c0],      &A[(size_t)(row_base + r0)      * Kdim + k0 + c0]);
    async_copy_b128(&As[buf][r0 + 32][c0], &A[(size_t)(row_base + r0 + 32) * Kdim + k0 + c0]);
    async_copy_b128(&Bs[buf][r0][c0],      &Bt[(size_t)(col_base + r0)      * Kdim + k0 + c0]);
    async_copy_b128(&Bs[buf][r0 + 32][c0], &Bt[(size_t)(col_base + r0 + 32) * Kdim + k0 + c0]);
  };

  v8f acc[2][2] = {};

  issue(0, k_lo);
  for (int kt = 0; kt < nk; ++kt) {
    const int cur = kt & 1;
    const bool more = (kt + 1 < nk);
    if (more) issue(1 - cur, k_lo + (kt + 1) * KSTEP);   // prefetch next tile
    if (more) asm volatile("s_wait_asynccnt 0x4" ::: "memory");  // cur's 4 done
    else      asm volatile("s_wait_asynccnt 0x0" ::: "memory");
    __syncthreads();

    const int kg   = (lane >> 4) << 3;  // A frag: K halves 0-7 / 8-15 (+16)
    const int koff = (lane >> 4) << 4;  // B frag: same K order as A frag
#pragma unroll
    for (int tm = 0; tm < 2; ++tm) {
      v16bf a;
      const int ar = wm * 32 + tm * 16 + (lane & 15);
#pragma unroll
      for (int i = 0; i < 8; ++i) {
        a[i]     = As[cur][ar][kg + i];
        a[i + 8] = As[cur][ar][16 + kg + i];
      }
#pragma unroll
      for (int tn = 0; tn < 2; ++tn) {
        v16bf b;
        const int bc = wn * 32 + tn * 16 + (lane & 15);
#pragma unroll
        for (int i = 0; i < 8; ++i) { b[i] = Bs[cur][bc][kg + i]; b[i + 8] = Bs[cur][bc][16 + kg + i]; }
        acc[tm][tn] = __builtin_amdgcn_wmma_f32_16x16x32_bf16(
            false, a, false, b, (short)0, acc[tm][tn], false, false);
      }
    }
    __syncthreads();  // all waves done reading cur before it is refilled
  }

  const int n  = lane & 15;
  const int mo = (lane >> 4) << 3;
#pragma unroll
  for (int tm = 0; tm < 2; ++tm) {
#pragma unroll
    for (int tn = 0; tn < 2; ++tn) {
      const int col = col_base + wn * 32 + tn * 16 + n;
      const float bv = bias ? bias[col] : 0.f;
#pragma unroll
      for (int r = 0; r < 8; ++r) {
        const int row = row_base + wm * 32 + tm * 16 + mo + r;
        float v = scale * acc[tm][tn][r] + bv;
        if (band_mask > 0) {
          int d = row - col; if (d < 0) d = -d;
          if (d >= band_mask) v = -__builtin_huge_valf();
        }
        if (Cf)  Cf[(size_t)row * Ndim + col] = v;
        if (Cb)  Cb[(size_t)row * Ndim + col] = (bhalf)v;
        if (CbT) CbT[(size_t)col * ldT + row] = (bhalf)v;
      }
    }
  }
}

// ---------------------------------------------------------------------------
// Banded row softmax, in place on att (f32), emitting att^T in bf16 so
// (att^T @ V) is a plain NT WMMA GEMM. One 256-thread block per row.
// Writes all N columns (zeros outside band; d_out is poisoned).
// ---------------------------------------------------------------------------
__global__ __launch_bounds__(256)
void softmax_band_kernel(float* __restrict__ att, bhalf* __restrict__ attT) {
  const int i   = blockIdx.x;
  const int tid = threadIdx.x;
  const int lo  = imax(i - (APER - 1), 0);
  const int hi  = imin(i + (APER - 1), N_FRAMES - 1);
  __shared__ float red[256];

  float m = -3.402823466e38f;
  for (int c = lo + tid; c <= hi; c += 256)
    m = fmaxf(m, att[(size_t)i * N_FRAMES + c]);
  red[tid] = m; __syncthreads();
  for (int s = 128; s > 0; s >>= 1) {
    if (tid < s) red[tid] = fmaxf(red[tid], red[tid + s]);
    __syncthreads();
  }
  const float rowmax = red[0]; __syncthreads();

  float s = 0.f;
  for (int c = lo + tid; c <= hi; c += 256)
    s += __expf(att[(size_t)i * N_FRAMES + c] - rowmax);
  red[tid] = s; __syncthreads();
  for (int t = 128; t > 0; t >>= 1) {
    if (tid < t) red[tid] += red[tid + t];
    __syncthreads();
  }
  const float inv = 1.0f / red[0];

  for (int c = tid; c < N_FRAMES; c += 256) {
    float v = 0.f;
    if (c >= lo && c <= hi)
      v = __expf(att[(size_t)i * N_FRAMES + c] - rowmax) * inv;
    att[(size_t)i * N_FRAMES + c]  = v;
    attT[(size_t)c * N_FRAMES + i] = (bhalf)v;
  }
}

// ---------------------------------------------------------------------------
// Bilinear grid-sample (align_corners=True), offsets from ofx/ofy; bf16 out.
// ---------------------------------------------------------------------------
__global__ void grid_sample_kernel(const float* __restrict__ x,
                                   const float* __restrict__ ofx,
                                   const float* __restrict__ ofy,
                                   bhalf* __restrict__ xnewb) {
  int idx = blockIdx.x * blockDim.x + threadIdx.x;
  if (idx >= N_FRAMES * M_FEAT) return;
  const int H = N_FRAMES, W = M_FEAT;
  const int i = idx >> 10;
  const int j = idx & (M_FEAT - 1);

  float gx = 2.0f * (float)i / (float)H - 1.0f + ofx[idx];
  gx = fminf(fmaxf(gx, -1.f), 1.f);
  float gy = 2.0f * (float)j / (float)W - 1.0f + ofy[idx];
  gy = fminf(fmaxf(gy, -1.f), 1.f);

  const float ix = (gx + 1.f) * (float)(W - 1) * 0.5f;
  const float iy = (gy + 1.f) * (float)(H - 1) * 0.5f;
  const float ix0 = floorf(ix), iy0 = floorf(iy);
  const float wx = ix - ix0, wy = iy - iy0;
  const int ix0i = imin(imax((int)ix0, 0), W - 1);
  const int ix1i = imin(ix0i + 1, W - 1);
  const int iy0i = imin(imax((int)iy0, 0), H - 1);
  const int iy1i = imin(iy0i + 1, H - 1);

  const float v00 = x[(size_t)iy0i * W + ix0i];
  const float v01 = x[(size_t)iy0i * W + ix1i];
  const float v10 = x[(size_t)iy1i * W + ix0i];
  const float v11 = x[(size_t)iy1i * W + ix1i];
  const float r = (1.f - wy) * ((1.f - wx) * v00 + wx * v01) +
                  wy * ((1.f - wx) * v10 + wx * v11);
  xnewb[idx] = (bhalf)r;
}

// ---------------------------------------------------------------------------
extern "C" void kernel_launch(void* const* d_in, const int* in_sizes, int n_in,
                              void* d_out, int out_size, void* d_ws, size_t ws_size,
                              hipStream_t stream) {
  (void)in_sizes; (void)n_in; (void)out_size; (void)ws_size;

  const float* x    = (const float*)d_in[0];
  const float* Wq   = (const float*)d_in[1];
  const float* Wk   = (const float*)d_in[2];
  const float* Wv   = (const float*)d_in[3];
  const float* Wofx = (const float*)d_in[4];
  const float* bofx = (const float*)d_in[5];
  const float* Wofy = (const float*)d_in[6];
  const float* bofy = (const float*)d_in[7];
  const float* Wout = (const float*)d_in[8];

  float* y_out   = (float*)d_out;                              // (N, M)
  float* att_out = (float*)d_out + (size_t)N_FRAMES * M_FEAT;  // (N, N)

  char* wsp = (char*)d_ws;
  auto alloc = [&](size_t bytes) -> void* {
    void* p = (void*)wsp;
    wsp += (bytes + 255) & ~(size_t)255;
    return p;
  };
  const size_t NM = (size_t)N_FRAMES * M_FEAT;
  const size_t MD = (size_t)M_FEAT * D_DIM;
  const size_t NN = (size_t)N_FRAMES * N_FRAMES;

  bhalf* xb    = (bhalf*)alloc(NM * 2);
  bhalf* Wqt   = (bhalf*)alloc(MD * 2);   // transposed bf16 weights (N x K)
  bhalf* Wkt   = (bhalf*)alloc(MD * 2);
  bhalf* Wvt   = (bhalf*)alloc(MD * 2);
  bhalf* Wofxt = (bhalf*)alloc(MD * 2);
  bhalf* Wofyt = (bhalf*)alloc(MD * 2);
  bhalf* Woutt = (bhalf*)alloc(MD * 2);
  bhalf* Qb    = (bhalf*)alloc(NM * 2);
  bhalf* Kb    = (bhalf*)alloc(NM * 2);   // row-major (N x D): already "B^T" for logits
  bhalf* Vt    = (bhalf*)alloc(NM * 2);   // transposed V (D x N)
  bhalf* xnewb = (bhalf*)alloc(NM * 2);
  bhalf* y1b   = (bhalf*)alloc(NM * 2);
  bhalf* attTb = (bhalf*)alloc(NN * 2);
  float* ofx   = (float*)alloc(NM * 4);
  float* ofy   = (float*)alloc(NM * 4);

  cvt_f32_bf16_kernel<<<(int)((NM + 255) / 256), 256, 0, stream>>>(x, xb, (int)NM);
  dim3 g_tr(D_DIM / 32, M_FEAT / 32);
  cvt_transpose_bf16_kernel<<<g_tr, 256, 0, stream>>>(Wq,   Wqt,   M_FEAT, D_DIM);
  cvt_transpose_bf16_kernel<<<g_tr, 256, 0, stream>>>(Wk,   Wkt,   M_FEAT, D_DIM);
  cvt_transpose_bf16_kernel<<<g_tr, 256, 0, stream>>>(Wv,   Wvt,   M_FEAT, D_DIM);
  cvt_transpose_bf16_kernel<<<g_tr, 256, 0, stream>>>(Wofx, Wofxt, D_DIM,  D_DIM);
  cvt_transpose_bf16_kernel<<<g_tr, 256, 0, stream>>>(Wofy, Wofyt, D_DIM,  D_DIM);
  cvt_transpose_bf16_kernel<<<g_tr, 256, 0, stream>>>(Wout, Woutt, D_DIM,  M_FEAT);

  dim3 g_nd(D_DIM / TILE, N_FRAMES / TILE);     // 16 x 64 tiles
  dim3 g_nn(N_FRAMES / TILE, N_FRAMES / TILE);  // 64 x 64 tiles

  // Q = x @ Wq  (bf16 out)
  gemm_nt_bf16_async_kernel<<<g_nd, 128, 0, stream>>>(
      xb, Wqt, nullptr, nullptr, Qb, nullptr, 0, D_DIM, M_FEAT, 1.0f, 0, 0);
  // ofx/ofy = Q @ Wof* + b  (f32 out)
  gemm_nt_bf16_async_kernel<<<g_nd, 128, 0, stream>>>(
      Qb, Wofxt, bofx, ofx, nullptr, nullptr, 0, D_DIM, D_DIM, 1.0f, 0, 0);
  gemm_nt_bf16_async_kernel<<<g_nd, 128, 0, stream>>>(
      Qb, Wofyt, bofy, ofy, nullptr, nullptr, 0, D_DIM, D_DIM, 1.0f, 0, 0);
  // x_new = grid_sample(x, gx, gy) -> bf16
  grid_sample_kernel<<<(int)((NM + 255) / 256), 256, 0, stream>>>(x, ofx, ofy, xnewb);
  // K = x_new @ Wk (row-major bf16) ; V = x_new @ Wv (transposed bf16)
  gemm_nt_bf16_async_kernel<<<g_nd, 128, 0, stream>>>(
      xnewb, Wkt, nullptr, nullptr, Kb, nullptr, 0, D_DIM, M_FEAT, 1.0f, 0, 0);
  gemm_nt_bf16_async_kernel<<<g_nd, 128, 0, stream>>>(
      xnewb, Wvt, nullptr, nullptr, nullptr, Vt, N_FRAMES, D_DIM, M_FEAT, 1.0f, 0, 0);
  // logits = SCALE * Q @ K^T (band-pruned tiles, -inf mask), f32 into d_out
  gemm_nt_bf16_async_kernel<<<g_nn, 128, 0, stream>>>(
      Qb, Kb, nullptr, att_out, nullptr, nullptr, 0, N_FRAMES, D_DIM, QSCALE, APER, 0);
  // att = softmax(logits) rows, in place; also att^T bf16
  softmax_band_kernel<<<N_FRAMES, 256, 0, stream>>>(att_out, attTb);
  // y1 = att^T @ V = attT @ Vt^T  (banded K-range)
  gemm_nt_bf16_async_kernel<<<g_nd, 128, 0, stream>>>(
      attTb, Vt, nullptr, nullptr, y1b, nullptr, 0, D_DIM, N_FRAMES, 1.0f, 0, APER);
  // y = y1 @ Wout  (f32 to d_out)
  gemm_nt_bf16_async_kernel<<<g_nd, 128, 0, stream>>>(
      y1b, Woutt, nullptr, y_out, nullptr, nullptr, 0, M_FEAT, D_DIM, 1.0f, 0, 0);
}
